// PCD_alignment_67946382622736
// MI455X (gfx1250) — matrix operand order repaired and checked by reading, not compiled, not run on hardware
//
#include <hip/hip_runtime.h>
#include <math.h>
#include <stdint.h>

typedef __attribute__((ext_vector_type(16))) _Float16 v16h;
typedef __attribute__((ext_vector_type(8)))  _Float16 v8h;
typedef __attribute__((ext_vector_type(8)))  float    v8f;
typedef __attribute__((ext_vector_type(4)))  unsigned int v4u;

#define NEG_SLOPE 0.2f
#define BATCH 4
#define C4 64

// ---------------------------------------------------------------------------
// CDNA5 async global->LDS staging (ASYNCcnt-tracked DMA, bypasses VGPRs).
// VDST = per-lane LDS byte offset, VADDR = per-lane 64-bit global address.
// Inline asm is used because the clang builtin arity differs across
// toolchains; the ISA mnemonic is stable.
// ---------------------------------------------------------------------------
__device__ __forceinline__ void async_stage_b128(unsigned lds_off, const void* g) {
  asm volatile("global_load_async_to_lds_b128 %0, %1, off"
               :: "v"(lds_off), "v"((unsigned long long)(uintptr_t)g)
               : "memory");
}
__device__ __forceinline__ void wait_async0() {
  asm volatile("s_wait_asynccnt 0x0" ::: "memory");
}

// ---------------------------------------------------------------------------
// Weight pack: fp32 (N,K) [OIHW flattened, or DCN (o,c,k) remap] -> fp16
// B-fragments in WMMA 32x16 per-lane layout, addressable as
//   Bp[(((kt*NT)+nt)*32 + lane)*8 + v]  (one dword = K=2v,2v+1 for this lane)
// lanes 0-15: K = kt*32 + 0..15, N = nt*16 + lane
// lanes 16-31: K = kt*32 + 16..31, N = nt*16 + (lane-16)
// ---------------------------------------------------------------------------
__global__ void pack_weights(const float* __restrict__ w, unsigned int* __restrict__ Bp,
                             int N, int K, int Kpad, int Npad, int mode, int C, int KPP) {
  int NT = Npad >> 4;
  long long total = (long long)(Kpad >> 5) * NT * 256;
  long long idx = (long long)blockIdx.x * blockDim.x + threadIdx.x;
  if (idx >= total) return;
  int v    = (int)(idx & 7);
  int lane = (int)((idx >> 3) & 31);
  long long rem = idx >> 8;
  int nt = (int)(rem % NT);
  int kt = (int)(rem / NT);
  int k0 = kt * 32 + ((lane >> 4) << 4) + 2 * v;
  int n  = nt * 16 + (lane & 15);
  union { _Float16 h[2]; unsigned int u; } pu;
  for (int j = 0; j < 2; ++j) {
    int k = k0 + j;
    float val = 0.f;
    if (k < K && n < N) {
      int flat = (mode == 0) ? k : ((k % C) * KPP + (k / C)); // DCN: kk = k9*C + c -> w[o][c][k9]
      val = w[(size_t)n * K + flat];
    }
    pu.h[j] = (_Float16)val;
  }
  Bp[idx] = pu.u;
}

// ---------------------------------------------------------------------------
// im2col (fp32 NCHW -> fp16 A panel [M][Kpad]); supports channel-concat of
// two sources, stride/pad, ks=1 or 3; zero-fills K padding.
// kk = (c*ks + ky)*ks + kx  matching OIHW weight flattening.
// ---------------------------------------------------------------------------
__global__ void im2col_f16(const float* __restrict__ s0, int C0,
                           const float* __restrict__ s1, int C1,
                           int H, int W, int OH, int OW, int stride, int pad, int ks,
                           int Kpad, _Float16* __restrict__ A) {
  long long total = (long long)BATCH * OH * OW * Kpad;
  long long idx = (long long)blockIdx.x * blockDim.x + threadIdx.x;
  if (idx >= total) return;
  int kk = (int)(idx % Kpad);
  long long m = idx / Kpad;
  int K = (C0 + C1) * ks * ks;
  float val = 0.f;
  if (kk < K) {
    int c  = kk / (ks * ks);
    int r  = kk % (ks * ks);
    int ky = r / ks, kx = r % ks;
    int ohw = (int)(m % (OH * OW));
    int b   = (int)(m / (OH * OW));
    int oy = ohw / OW, ox = ohw % OW;
    int iy = oy * stride - pad + ky;
    int ix = ox * stride - pad + kx;
    if (iy >= 0 && iy < H && ix >= 0 && ix < W) {
      const float* src = (c < C0) ? (s0 + ((size_t)b * C0 + c) * H * W)
                                  : (s1 + ((size_t)b * C1 + (c - C0)) * H * W);
      val = src[iy * W + ix];
    }
  }
  A[idx] = (_Float16)val;
}

// ---------------------------------------------------------------------------
// DCN-v2 sampler: bilinear gather (zero outside) * sigmoid(mask) into the
// same fp16 A panel, K layout kk = k9*64 + c  (Kpad = 576).
// ---------------------------------------------------------------------------
__global__ void dcn_sample(const float* __restrict__ f,
                           const float* __restrict__ offp, int offC,
                           const float* __restrict__ maskp, int maskC, int maskC0,
                           int H, int W, _Float16* __restrict__ A) {
  long long total = (long long)BATCH * H * W * 9;
  long long idx = (long long)blockIdx.x * blockDim.x + threadIdx.x;
  if (idx >= total) return;
  int k = (int)(idx % 9);
  long long m = idx / 9;
  size_t HWs = (size_t)H * W;
  int hw = (int)(m % HWs);
  int b  = (int)(m / HWs);
  int yy = hw / W, xx = hw % W;
  float dy = offp[((size_t)b * offC + 2 * k) * HWs + hw];
  float dx = offp[((size_t)b * offC + 2 * k + 1) * HWs + hw];
  float mk = maskp[((size_t)b * maskC + maskC0 + k) * HWs + hw];
  float sg = 1.f / (1.f + expf(-mk));
  float py = dy + (float)yy + (float)(k / 3 - 1);
  float px = dx + (float)xx + (float)(k % 3 - 1);
  float y0f = floorf(py), x0f = floorf(px);
  int y0 = (int)y0f, x0 = (int)x0f;
  float wy = py - y0f, wx = px - x0f;
  int y1 = y0 + 1, x1 = x0 + 1;
  float v00m = (y0 >= 0 && y0 < H && x0 >= 0 && x0 < W) ? 1.f : 0.f;
  float v01m = (y0 >= 0 && y0 < H && x1 >= 0 && x1 < W) ? 1.f : 0.f;
  float v10m = (y1 >= 0 && y1 < H && x0 >= 0 && x0 < W) ? 1.f : 0.f;
  float v11m = (y1 >= 0 && y1 < H && x1 >= 0 && x1 < W) ? 1.f : 0.f;
  int y0c = min(max(y0, 0), H - 1), y1c = min(max(y1, 0), H - 1);
  int x0c = min(max(x0, 0), W - 1), x1c = min(max(x1, 0), W - 1);
  float w00 = (1.f - wy) * (1.f - wx) * v00m * sg;
  float w01 = (1.f - wy) * wx * v01m * sg;
  float w10 = wy * (1.f - wx) * v10m * sg;
  float w11 = wy * wx * v11m * sg;
  _Float16* dst = A + (size_t)m * 576 + (size_t)k * C4;
  const float* base = f + (size_t)b * C4 * HWs;
  for (int c = 0; c < C4; ++c) {
    const float* pc = base + (size_t)c * HWs;
    float s = pc[y0c * W + x0c] * w00 + pc[y0c * W + x1c] * w01 +
              pc[y1c * W + x0c] * w10 + pc[y1c * W + x1c] * w11;
    dst[c] = (_Float16)s;
  }
}

// ---------------------------------------------------------------------------
// WMMA implicit GEMM: out[b, n, hw] = A[m][:] . B[:][n] + bias (+ lrelu)
// block = 128 threads (4 wave32): 32 M-rows x 64 N-cols per block.
// Each wave owns one 16-wide N-tile and TWO 16-row M-subtiles: a single
// B fragment feeds two back-to-back v_wmma ops (register-level B reuse).
// A tiles (32x32 fp16) are double-buffered in LDS and staged with the
// CDNA5 async global->LDS DMA (ASYNCcnt); ONE barrier per K-step.
// ---------------------------------------------------------------------------
__global__ void wmma_gemm_f16(const _Float16* __restrict__ A,
                              const unsigned int* __restrict__ Bp,
                              const float* __restrict__ bias,
                              float* __restrict__ out,
                              int M, int Kpad, int N, int Npad, int OHW, int do_lrelu) {
  __shared__ __align__(16) _Float16 As[2][32 * 32];
  int lane = threadIdx.x & 31;
  int wave = threadIdx.x >> 5;
  int m0 = blockIdx.x * 32;
  int NT = Npad >> 4;
  int nt = blockIdx.y * 4 + wave;
  bool active = (nt < NT);
  int arow = lane & 15;
  int kb = (lane >> 4) * 16;         // A frag: K = kb..kb+15
  int srow = threadIdx.x >> 2;       // cooperative stage: 32 rows x 4 x 16B
  int sseg = threadIdx.x & 3;
  int ksteps = Kpad >> 5;
  v8f acc0 = {};
  v8f acc1 = {};

  // Per-lane LDS byte offsets for the two staging buffers (flat LDS address
  // truncates to addr[31:0] = wave-relative LDS byte offset).
  unsigned lds0 = (unsigned)(uintptr_t)&As[0][srow * 32 + sseg * 8];
  unsigned lds1 = (unsigned)(uintptr_t)&As[1][srow * 32 + sseg * 8];
  const _Float16* grow = A + (size_t)(m0 + srow) * Kpad + sseg * 8;

  // stage tile 0 via async DMA
  async_stage_b128(lds0, grow);
  wait_async0();
  __syncthreads();

  for (int kt = 0; kt < ksteps; ++kt) {
    int cur = kt & 1;
    // stage next tile into the other buffer (DMA overlaps compute below)
    if (kt + 1 < ksteps) {
      async_stage_b128(cur ? lds0 : lds1, grow + (kt + 1) * 32);
      if (kt + 2 < ksteps)
        __builtin_prefetch(grow + (kt + 2) * 32, 0, 1);
    }
    if (active) {
      union { v8h h8[2]; v16h h16; } a0, a1;
      a0.h8[0] = *(const v8h*)&As[cur][arow * 32 + kb];
      a0.h8[1] = *(const v8h*)&As[cur][arow * 32 + kb + 8];
      a1.h8[0] = *(const v8h*)&As[cur][(arow + 16) * 32 + kb];
      a1.h8[1] = *(const v8h*)&As[cur][(arow + 16) * 32 + kb + 8];
      const unsigned int* bp = Bp + (((size_t)kt * NT + nt) * 32 + lane) * 8;
      union { v4u u[2]; v16h h16; } bu;
      bu.u[0] = *(const v4u*)bp;
      bu.u[1] = *(const v4u*)(bp + 4);
      acc0 = __builtin_amdgcn_wmma_f32_16x16x32_f16(
          false, a0.h16, false, bu.h16, (short)0, acc0, false, false);
      acc1 = __builtin_amdgcn_wmma_f32_16x16x32_f16(
          false, a1.h16, false, bu.h16, (short)0, acc1, false, false);
    }
    // drain the DMA, then one barrier: next tile staged before next compute,
    // and current tile consumed before it is overwritten at step kt+1.
    if (kt + 1 < ksteps) wait_async0();
    __syncthreads();
  }

  if (active) {
    int n = nt * 16 + (lane & 15);
    if (n < N) {
      float bv = bias ? bias[n] : 0.f;
      int mbase = m0 + ((lane >> 4) ? 8 : 0);
      for (int r = 0; r < 8; ++r) {
        int m = mbase + r;
        float v = acc0[r] + bv;
        if (do_lrelu) v = (v >= 0.f) ? v : NEG_SLOPE * v;
        int b = m / OHW, hw = m % OHW;
        out[((size_t)b * N + n) * OHW + hw] = v;

        int m2 = m + 16;
        float v2 = acc1[r] + bv;
        if (do_lrelu) v2 = (v2 >= 0.f) ? v2 : NEG_SLOPE * v2;
        int b2 = m2 / OHW, hw2 = m2 % OHW;
        out[((size_t)b2 * N + n) * OHW + hw2] = v2;
      }
    }
  }
}

// ---------------------------------------------------------------------------
// Bilinear 2x upsample, reference index arithmetic.
// ---------------------------------------------------------------------------
__global__ void up2x_k(const float* __restrict__ x, float* __restrict__ y,
                       int C, int H, int W) {
  int H2 = 2 * H, W2 = 2 * W;
  long long total = (long long)BATCH * C * H2 * W2;
  long long idx = (long long)blockIdx.x * blockDim.x + threadIdx.x;
  if (idx >= total) return;
  int x2 = (int)(idx % W2);
  int y2 = (int)((idx / W2) % H2);
  int c  = (int)((idx / ((long long)W2 * H2)) % C);
  int b  = (int)(idx / ((long long)W2 * H2 * C));
  float sy = (float)y2 * ((float)(H - 1) / (float)(2 * H - 1));
  float sx = (float)x2 * ((float)(W - 1) / (float)(2 * W - 1));
  int i0 = min(max((int)floorf(sy), 0), H - 2);
  int j0 = min(max((int)floorf(sx), 0), W - 2);
  float ty = sy - (float)i0;
  float tx = sx - (float)j0;
  const float* p = x + ((size_t)b * C + c) * H * W;
  float v00 = p[i0 * W + j0],       v01 = p[i0 * W + j0 + 1];
  float v10 = p[(i0 + 1) * W + j0], v11 = p[(i0 + 1) * W + j0 + 1];
  float xh0 = v00 * (1.f - ty) + v10 * ty;
  float xh1 = v01 * (1.f - ty) + v11 * ty;
  y[idx] = xh0 * (1.f - tx) + xh1 * tx;
}

__global__ void slice_ch(const float* __restrict__ src, float* __restrict__ dst,
                         int inC, int c0, int cn, int HW) {
  long long total = (long long)BATCH * cn * HW;
  long long idx = (long long)blockIdx.x * blockDim.x + threadIdx.x;
  if (idx >= total) return;
  int hw = (int)(idx % HW);
  int c  = (int)((idx / HW) % cn);
  int b  = (int)(idx / ((long long)HW * cn));
  dst[idx] = src[((size_t)b * inC + c0 + c) * HW + hw];
}

__global__ void zero_scalar(float* p) { *p = 0.f; }

__global__ void reduce_sum(const float* __restrict__ x, long long n, float* acc) {
  __shared__ float s[256];
  long long i = (long long)blockIdx.x * blockDim.x + threadIdx.x;
  long long stride = (long long)gridDim.x * blockDim.x;
  float v = 0.f;
  for (; i < n; i += stride) v += x[i];
  s[threadIdx.x] = v;
  __syncthreads();
  for (int o = 128; o > 0; o >>= 1) {
    if ((int)threadIdx.x < o) s[threadIdx.x] += s[threadIdx.x + o];
    __syncthreads();
  }
  if (threadIdx.x == 0) atomicAdd(acc, s[0]);
}

// ---------------------------------------------------------------------------
// Host orchestration
// ---------------------------------------------------------------------------
extern "C" void kernel_launch(void* const* d_in, const int* in_sizes, int n_in,
                              void* d_out, int out_size, void* d_ws, size_t ws_size,
                              hipStream_t stream) {
  (void)in_sizes; (void)ws_size;
  const float* P[64];
  for (int i = 0; i < n_in && i < 64; ++i) P[i] = (const float*)d_in[i];
  const float* c1 = P[0];
  const float* a1 = P[1];

  char* ws = (char*)d_ws;
  size_t off = 0;
  auto alloc = [&](size_t bytes) -> char* {
    char* p = ws + off;
    off += (bytes + 255) & ~(size_t)255;
    return p;
  };
  _Float16*     Acol = (_Float16*)alloc((size_t)65536 * 576 * 2); // max A panel
  unsigned int* Bp   = (unsigned int*)alloc(256 * 1024);
  float* C2  = (float*)alloc((size_t)BATCH * C4 * 64 * 64 * 4);
  float* A2  = (float*)alloc((size_t)BATCH * C4 * 64 * 64 * 4);
  float* C3  = (float*)alloc((size_t)BATCH * C4 * 32 * 32 * 4);
  float* A3  = (float*)alloc((size_t)BATCH * C4 * 32 * 32 * 4);
  float* F      = (float*)alloc((size_t)BATCH * C4 * 128 * 128 * 4);
  float* OWb    = (float*)alloc((size_t)BATCH * 27 * 128 * 128 * 4);
  float* OFFT   = (float*)alloc((size_t)BATCH * 18 * 128 * 128 * 4);
  float* OFFB   = (float*)alloc((size_t)BATCH * 18 * 128 * 128 * 4);
  float* UPOFF  = (float*)alloc((size_t)BATCH * 18 * 128 * 128 * 4);
  float* FEAT   = (float*)alloc((size_t)BATCH * C4 * 128 * 128 * 4);
  float* UPFEAT = (float*)alloc((size_t)BATCH * C4 * 128 * 128 * 4);
  float* TMP    = (float*)alloc((size_t)BATCH * C4 * 128 * 128 * 4);

  float* total_ptr = (float*)d_out + (out_size - 1);

  auto gemm = [&](const _Float16* Ac, const float* bias, float* out,
                  int M, int Kpad, int N, int do_lrelu, int OHW) {
    int Npad = (N + 15) & ~15;
    dim3 g(M / 32, (Npad + 63) / 64);
    wmma_gemm_f16<<<g, 128, 0, stream>>>(Ac, Bp, bias, out, M, Kpad, N, Npad, OHW, do_lrelu);
  };
  auto conv = [&](const float* s0, int C0, const float* s1, int C1,
                  int H, int W, int stride, int pad, int ks,
                  const float* w, const float* bias, int N, float* out, int lre) {
    int OH = (H + 2 * pad - ks) / stride + 1;
    int OW = (W + 2 * pad - ks) / stride + 1;
    int K = (C0 + C1) * ks * ks;
    int Kpad = (K + 31) & ~31;
    int Npad = (N + 15) & ~15;
    long long pt = (long long)(Kpad >> 5) * (Npad >> 4) * 256;
    pack_weights<<<(int)((pt + 255) / 256), 256, 0, stream>>>(w, Bp, N, K, Kpad, Npad, 0, 0, 0);
    long long it = (long long)BATCH * OH * OW * Kpad;
    im2col_f16<<<(int)((it + 255) / 256), 256, 0, stream>>>(s0, C0, s1, C1, H, W, OH, OW,
                                                            stride, pad, ks, Kpad, Acol);
    gemm(Acol, bias, out, BATCH * OH * OW, Kpad, N, lre, OH * OW);
  };

  zero_scalar<<<1, 1, 0, stream>>>(total_ptr);

  // Feature pyramid (shared weights per level)
  conv(c1, C4, nullptr, 0, 128, 128, 2, 1, 3, P[2], P[3], C4, C2, 0);
  conv(a1, C4, nullptr, 0, 128, 128, 2, 1, 3, P[2], P[3], C4, A2, 0);
  conv(C2, C4, nullptr, 0, 64, 64, 2, 1, 3, P[4], P[5], C4, C3, 0);
  conv(A2, C4, nullptr, 0, 64, 64, 2, 1, 3, P[4], P[5], C4, A3, 0);

  struct Lvl { int H; const float *cen, *ali; int oc1, oc2, ow, dcn, fus, fc, otc, ftc; };
  Lvl lv[3] = {
      {32,  C3, A3, 34, 36, 38, 40, -1, -1, 42, 44},  // level 3
      {64,  C2, A2, 18, 20, 22, 24, 26, 28, 30, 32},  // level 2
      {128, c1, a1,  6,  8, 10, 12, 14, 16, -1, -1},  // level 1
  };

  for (int li = 0; li < 3; ++li) {
    const Lvl& l = lv[li];
    int H = l.H, W = l.H, HW = H * W, M = BATCH * HW;

    conv(l.cen, C4, l.ali, C4, H, W, 1, 0, 1, P[l.oc1], P[l.oc1 + 1], C4, F, 0);
    conv(F, C4, nullptr, 0, H, W, 1, 1, 3, P[l.oc2], P[l.oc2 + 1], C4, F, 1);
    conv(F, C4, nullptr, 0, H, W, 1, 1, 3, P[l.ow], P[l.ow + 1], 27, OWb, 0);

    { long long t = (long long)BATCH * 18 * HW;
      slice_ch<<<(int)((t + 255) / 256), 256, 0, stream>>>(OWb, OFFT, 27, 0, 18, HW); }

    float* offcur = OFFT;
    if (l.fus >= 0) {
      conv(OFFT, 18, UPOFF, 18, H, W, 1, 0, 1, P[l.fus], P[l.fus + 1], 18, OFFB, 0);
      offcur = OFFB;
    }

    // Deformable conv v2 as sampler + WMMA GEMM (K = 9*64 = 576)
    {
      long long pt = (long long)(576 / 32) * (64 / 16) * 256;
      pack_weights<<<(int)((pt + 255) / 256), 256, 0, stream>>>(P[l.dcn], Bp, C4, 576, 576, 64, 1, C4, 9);
      long long st = (long long)M * 9;
      dcn_sample<<<(int)((st + 255) / 256), 256, 0, stream>>>(F, offcur, 18, OWb, 27, 18, H, W, Acol);
      gemm(Acol, P[l.dcn + 1], FEAT, M, 576, C4, 0, HW);
    }

    float* featcur = FEAT;
    if (l.fc >= 0) {
      float* dst = (li == 2) ? (float*)d_out : FEAT;
      conv(FEAT, C4, UPFEAT, C4, H, W, 1, 0, 1, P[l.fc], P[l.fc + 1], C4, dst, 1);
      featcur = dst;
    }

    if (l.otc >= 0) {
      long long t = (long long)BATCH * 18 * 4 * HW;
      up2x_k<<<(int)((t + 255) / 256), 256, 0, stream>>>(offcur, TMP, 18, H, W);
      conv(TMP, 18, nullptr, 0, 2 * H, 2 * W, 1, 1, 3, P[l.otc], P[l.otc + 1], 18, UPOFF, 0);
      t = (long long)BATCH * C4 * 4 * HW;
      up2x_k<<<(int)((t + 255) / 256), 256, 0, stream>>>(featcur, TMP, C4, H, W);
      conv(TMP, C4, nullptr, 0, 2 * H, 2 * W, 1, 1, 3, P[l.ftc], P[l.ftc + 1], C4, UPFEAT, 0);
    }

    { long long t = (long long)BATCH * 18 * HW;
      reduce_sum<<<256, 256, 0, stream>>>(offcur, t, total_ptr); }
  }
}
